// MyModel_83399674954386
// MI455X (gfx1250) — matrix operand order repaired and checked by reading, not compiled, not run on hardware
//
#include <hip/hip_runtime.h>
#include <hip/hip_bf16.h>

#define NN   50000
#define EE   800000
#define INF_ 256
#define DD   64
#define HH   4
#define NEGS 0.2f

typedef __attribute__((ext_vector_type(16))) __bf16 v16bf;
typedef __attribute__((ext_vector_type(8)))  __bf16 v8bf;
typedef __attribute__((ext_vector_type(8)))  float  v8f;

// ---------- helpers ----------
__device__ __forceinline__ unsigned fenc(float f) {
    unsigned u = __float_as_uint(f);
    return (u & 0x80000000u) ? ~u : (u | 0x80000000u);
}
__device__ __forceinline__ float fdec(unsigned u) {
    unsigned b = (u & 0x80000000u) ? (u & 0x7FFFFFFFu) : ~u;
    return __uint_as_float(b);
}
__device__ __forceinline__ float sigf(float x) { return 1.f / (1.f + expf(-x)); }

// ---------- conversions ----------
__global__ void f32_to_bf16_kernel(const float* __restrict__ in, __bf16* __restrict__ out, long long n) {
    long long i = (long long)blockIdx.x * blockDim.x + threadIdx.x;
    if (i < n) out[i] = (__bf16)in[i];
}

// W (K x Nout, row-major) -> Wt (Nout x K, row-major, bf16)
__global__ void transpose_to_bf16_kernel(const float* __restrict__ W, __bf16* __restrict__ Wt,
                                         int K, int Nout) {
    int i = blockIdx.x * blockDim.x + threadIdx.x;
    if (i >= K * Nout) return;
    int k = i / Nout, n = i % Nout;
    Wt[(size_t)n * K + k] = (__bf16)W[i];
}

__global__ void elu_to_bf16_kernel(const float* __restrict__ in, __bf16* __restrict__ out, long long n) {
    long long i = (long long)blockIdx.x * blockDim.x + threadIdx.x;
    if (i < n) { float v = in[i]; out[i] = (__bf16)(v > 0.f ? v : expm1f(v)); }
}

__global__ void elu_inplace_kernel(float* __restrict__ x, long long n) {
    long long i = (long long)blockIdx.x * blockDim.x + threadIdx.x;
    if (i < n) { float v = x[i]; x[i] = v > 0.f ? v : expm1f(v); }
}

// ---------- WMMA GEMM: C[M x Nout] = A[M x K](bf16) @ Wt[Nout x K](bf16)^T + bias ----------
// one wave (32 lanes) computes a 16x64 output strip; grid=(M/16, Nout/64)
__global__ void __launch_bounds__(32)
wmma_gemm_bf16_kernel(const __bf16* __restrict__ A, const __bf16* __restrict__ Wt,
                      const float* __restrict__ bias, float* __restrict__ C,
                      int K, int Nout) {
    const int lane = threadIdx.x;
    const int lh = lane >> 4;        // lane half (0/1)
    const int ll = lane & 15;
    const int m0 = blockIdx.x * 16;
    const int n0 = blockIdx.y * 64;

    const __bf16* arow = A + (size_t)(m0 + ll) * K;
    const int aklo = lh * 16;        // A: lanes0-15 K=0..15 ; lanes16-31 K=16..31
    const __bf16* b0 = Wt + (size_t)(n0 + ll) * K + lh * 16;  // B: 16 contiguous K per lane
    const __bf16* b1 = b0 + (size_t)16 * K;
    const __bf16* b2 = b0 + (size_t)32 * K;
    const __bf16* b3 = b0 + (size_t)48 * K;

    v8f acc0 = {}, acc1 = {}, acc2 = {}, acc3 = {};
    for (int k0 = 0; k0 < K; k0 += 32) {
        v8bf alo = *(const v8bf*)(arow + k0 + aklo);
        v8bf ahi = *(const v8bf*)(arow + k0 + 8 + aklo);
        v16bf a;
#pragma unroll
        for (int i = 0; i < 8; ++i) { a[i] = alo[i]; a[i + 8] = ahi[i]; }
        v16bf vb0 = *(const v16bf*)(b0 + k0);
        v16bf vb1 = *(const v16bf*)(b1 + k0);
        v16bf vb2 = *(const v16bf*)(b2 + k0);
        v16bf vb3 = *(const v16bf*)(b3 + k0);
        acc0 = __builtin_amdgcn_wmma_f32_16x16x32_bf16(false, a, false, vb0, (short)0, acc0, false, false);
        acc1 = __builtin_amdgcn_wmma_f32_16x16x32_bf16(false, a, false, vb1, (short)0, acc1, false, false);
        acc2 = __builtin_amdgcn_wmma_f32_16x16x32_bf16(false, a, false, vb2, (short)0, acc2, false, false);
        acc3 = __builtin_amdgcn_wmma_f32_16x16x32_bf16(false, a, false, vb3, (short)0, acc3, false, false);
    }
    const int rbase = lh * 8;        // C/D layout: lanes0-15 M=r, lanes16-31 M=8+r
    v8f accs[4] = {acc0, acc1, acc2, acc3};
#pragma unroll
    for (int j = 0; j < 4; ++j) {
        int col = n0 + 16 * j + ll;
        float bv = bias[col];
#pragma unroll
        for (int r = 0; r < 8; ++r)
            C[(size_t)(m0 + rbase + r) * Nout + col] = accs[j][r] + bv;
    }
}

// ---------- edge phase ----------
__global__ void edge_score_kernel(const float* __restrict__ fs, const float* __restrict__ fd,
                                  const int* __restrict__ src, const int* __restrict__ dst,
                                  const float* __restrict__ attn, float* __restrict__ score,
                                  unsigned* __restrict__ emax, long long EH, int Hn, int Dh) {
    long long i = (long long)blockIdx.x * blockDim.x + threadIdx.x;
    if (i >= EH) return;
    int e = (int)(i / Hn), h = (int)(i % Hn);
    int F = Hn * Dh;
    const float* ps = fs + (size_t)src[e] * F + h * Dh;
    const float* pd = fd + (size_t)dst[e] * F + h * Dh;
    const float* pa = attn + h * Dh;
    float s = 0.f;
    for (int d = 0; d < Dh; ++d) {
        float v = ps[d] + pd[d];
        v = v > 0.f ? v : NEGS * v;
        s += v * pa[d];
    }
    score[i] = s;
    atomicMax(&emax[(size_t)dst[e] * Hn + h], fenc(s));
}

__global__ void edge_exp_kernel(float* __restrict__ score, const int* __restrict__ dst,
                                const unsigned* __restrict__ emax, float* __restrict__ ssum,
                                long long EH, int Hn) {
    long long i = (long long)blockIdx.x * blockDim.x + threadIdx.x;
    if (i >= EH) return;
    int e = (int)(i / Hn), h = (int)(i % Hn);
    float m = fdec(emax[(size_t)dst[e] * Hn + h]);
    float ex = expf(score[i] - m);
    score[i] = ex;
    atomicAdd(&ssum[(size_t)dst[e] * Hn + h], ex);
}

// ---------- CSR build ----------
__global__ void count_deg_kernel(const int* __restrict__ dst, int* __restrict__ deg, int E) {
    int i = blockIdx.x * blockDim.x + threadIdx.x;
    if (i < E) atomicAdd(&deg[dst[i]], 1);
}

__global__ void __launch_bounds__(1024)
exscan_kernel(const int* __restrict__ in, int* __restrict__ out, int n) {
    __shared__ int buf[1024];
    __shared__ int carry_s;
    const int t = threadIdx.x;
    if (t == 0) carry_s = 0;
    __syncthreads();
    for (int base = 0; base < n; base += 1024) {
        int i = base + t;
        int v = (i < n) ? in[i] : 0;
        buf[t] = v;
        __syncthreads();
        for (int off = 1; off < 1024; off <<= 1) {
            int add = (t >= off) ? buf[t - off] : 0;
            __syncthreads();
            buf[t] += add;
            __syncthreads();
        }
        int incl = buf[t];
        int c = carry_s;
        if (i < n) out[i] = c + incl - v;
        __syncthreads();
        if (t == 1023) carry_s = c + incl;
        __syncthreads();
    }
    if (t == 0) out[n] = carry_s;
}

__global__ void scatter_edges_kernel(const int* __restrict__ dst, int* __restrict__ cursor,
                                     int* __restrict__ eidx, int E) {
    int i = blockIdx.x * blockDim.x + threadIdx.x;
    if (i < E) { int p = atomicAdd(&cursor[dst[i]], 1); eidx[p] = i; }
}

// ---------- aggregation (gather, no atomics): block per node, thread per (h,d) ----------
__global__ void aggregate_kernel(const float* __restrict__ ex, const float* __restrict__ ssum,
                                 const float* __restrict__ fs, const int* __restrict__ src,
                                 const int* __restrict__ rowptr, const int* __restrict__ eidx,
                                 float* __restrict__ out, int Hn, int Dh) {
    const int F = Hn * Dh;
    const int nID = blockIdx.x;
    const int t = threadIdx.x;          // 0..F-1
    const int h = t / Dh;
    const int beg = rowptr[nID], end = rowptr[nID + 1];
    float sv = ssum[(size_t)nID * Hn + h];
    float inv = sv > 0.f ? 1.f / sv : 0.f;
    float acc = 0.f;
    for (int k = beg; k < end; ++k) {
        int e = eidx[k];
        acc += ex[(size_t)e * Hn + h] * fs[(size_t)src[e] * F + t];
    }
    out[(size_t)nID * F + t] = acc * inv;
}

// ---------- Set2Set ----------
__global__ void __launch_bounds__(256)
lstm_kernel(const float* __restrict__ q_star, float* __restrict__ hs, float* __restrict__ cs,
            const float* Wi0, const float* Wh0, const float* bi0, const float* bh0,
            const float* Wi1, const float* Wh1, const float* bi1, const float* bh1,
            const float* Wi2, const float* Wh2, const float* bi2, const float* bh2,
            float* __restrict__ qout) {
    __shared__ float xbuf[128];
    __shared__ float gbuf[256];
    const int t = threadIdx.x;
    if (t < 128) xbuf[t] = q_star[t];
    __syncthreads();
    const float* Wi[3] = {Wi0, Wi1, Wi2};
    const float* Wh[3] = {Wh0, Wh1, Wh2};
    const float* bi[3] = {bi0, bi1, bi2};
    const float* bh[3] = {bh0, bh1, bh2};
#pragma unroll
    for (int l = 0; l < 3; ++l) {
        const int in_sz = (l == 0) ? 128 : 64;
        float g = bi[l][t] + bh[l][t];
        for (int k = 0; k < in_sz; ++k) g += xbuf[k] * Wi[l][(size_t)k * 256 + t];
        const float* hl = hs + l * 64;
        for (int k = 0; k < 64; ++k) g += hl[k] * Wh[l][(size_t)k * 256 + t];
        gbuf[t] = g;
        __syncthreads();
        if (t < 64) {
            float ig = sigf(gbuf[t]);
            float fg = sigf(gbuf[64 + t]);
            float gg = tanhf(gbuf[128 + t]);
            float og = sigf(gbuf[192 + t]);
            float c2 = fg * cs[l * 64 + t] + ig * gg;
            float h2 = og * tanhf(c2);
            cs[l * 64 + t] = c2;
            hs[l * 64 + t] = h2;
            xbuf[t] = h2;
        }
        __syncthreads();
    }
    if (t < 64) qout[t] = xbuf[t];
}

__global__ void s2s_logits_kernel(const float* __restrict__ feat, const float* __restrict__ q,
                                  float* __restrict__ logits, unsigned* __restrict__ gmax, int n) {
    __shared__ float qs[64];
    int t = threadIdx.x;
    if (t < 64) qs[t] = q[t];
    __syncthreads();
    int i = blockIdx.x * blockDim.x + t;
    if (i >= n) return;
    const float* f = feat + (size_t)i * 64;
    float s = 0.f;
    for (int d = 0; d < 64; ++d) s += f[d] * qs[d];
    logits[i] = s;
    atomicMax(gmax, fenc(s));
}

__global__ void s2s_exp_kernel(float* __restrict__ logits, const unsigned* __restrict__ gmax,
                               float* __restrict__ gsum, int n) {
    int i = blockIdx.x * blockDim.x + threadIdx.x;
    if (i >= n) return;
    float m = fdec(*gmax);
    float e = expf(logits[i] - m);
    logits[i] = e;
    atomicAdd(gsum, e);
}

__global__ void __launch_bounds__(256)
s2s_reduce_kernel(const float* __restrict__ feat, const float* __restrict__ ex,
                  float* __restrict__ r, int n) {
    __shared__ float sd[256];
    int t = threadIdx.x;
    int d = t & 63, g = t >> 6;
    float acc = 0.f;
    for (int i = blockIdx.x * 4 + g; i < n; i += gridDim.x * 4)
        acc += ex[i] * feat[(size_t)i * 64 + d];
    sd[t] = acc;
    __syncthreads();
    if (g == 0) {
        float tot = sd[d] + sd[64 + d] + sd[128 + d] + sd[192 + d];
        atomicAdd(&r[d], tot);
    }
}

__global__ void s2s_qstar_kernel(const float* __restrict__ q, const float* __restrict__ r,
                                 const float* __restrict__ gsum, float* __restrict__ q_star) {
    int t = threadIdx.x; // 128
    if (t < 64) q_star[t] = q[t];
    else        q_star[t] = r[t - 64] / gsum[0];
}

__global__ void copy_out_kernel(const float* __restrict__ in, float* __restrict__ out, int n) {
    int i = threadIdx.x;
    if (i < n) out[i] = in[i];
}

// ---------- launch ----------
extern "C" void kernel_launch(void* const* d_in, const int* in_sizes, int n_in,
                              void* d_out, int out_size, void* d_ws, size_t ws_size,
                              hipStream_t stream) {
    const float* feat  = (const float*)d_in[0];
    const int*   src   = (const int*)d_in[1];
    const int*   dst   = (const int*)d_in[2];
    const float* W1s   = (const float*)d_in[3];
    const float* b1s   = (const float*)d_in[4];
    const float* W1d   = (const float*)d_in[5];
    const float* b1d   = (const float*)d_in[6];
    const float* attn1 = (const float*)d_in[7];
    const float* W2s   = (const float*)d_in[8];
    const float* b2s   = (const float*)d_in[9];
    const float* W2d   = (const float*)d_in[10];
    const float* b2d   = (const float*)d_in[11];
    const float* attn2 = (const float*)d_in[12];
    const float* lW[3][4]; // Wi, Wh, bi, bh
    for (int l = 0; l < 3; ++l)
        for (int j = 0; j < 4; ++j)
            lW[l][j] = (const float*)d_in[13 + l * 4 + j];

    char* ws = (char*)d_ws;
    size_t off = 0;
    auto alloc = [&](size_t bytes) -> void* {
        void* p = ws + off;
        off = (off + bytes + 255) & ~(size_t)255;
        return p;
    };

    __bf16* bfA    = (__bf16*)alloc((size_t)NN * INF_ * 2);   // feat_bf16, later h1_bf16
    __bf16* w1s_t  = (__bf16*)alloc((size_t)INF_ * INF_ * 2);
    __bf16* w1d_t  = (__bf16*)alloc((size_t)INF_ * INF_ * 2);
    __bf16* w2s_t  = (__bf16*)alloc((size_t)DD * INF_ * 2);
    __bf16* w2d_t  = (__bf16*)alloc((size_t)DD * INF_ * 2);
    float*  R1     = (float*)alloc((size_t)NN * INF_ * 4);    // fs1; later fs2 / h2 / score2
    float*  R2     = (float*)alloc((size_t)NN * INF_ * 4);    // fd1; later agg1 out / fd2
    float*  score1 = (float*)alloc((size_t)EE * HH * 4);      // later set2set logits
    unsigned* msum = (unsigned*)alloc((size_t)NN * HH * 4);
    float*  ssum   = (float*)alloc((size_t)NN * HH * 4);
    int*    deg    = (int*)alloc((size_t)(NN + 1) * 4);
    int*    rowptr = (int*)alloc((size_t)(NN + 1) * 4);
    int*    cursor = (int*)alloc((size_t)(NN + 1) * 4);
    int*    eidx   = (int*)alloc((size_t)EE * 4);
    float*  q_star = (float*)alloc(512);
    float*  hsbuf  = (float*)alloc(3 * 64 * 4);
    float*  csbuf  = (float*)alloc(3 * 64 * 4);
    float*  qv     = (float*)alloc(256);
    float*  rv     = (float*)alloc(256);
    unsigned* gmax = (unsigned*)alloc(256);
    float*  gsum   = (float*)alloc(256);

    float* fs1 = R1;
    float* fd1 = R2;
    float* agg1 = R2;                       // fd1 dead by aggregation time
    float* fs2 = R1;                        // fs1 dead after aggregation 1
    float* h2  = R1 + (size_t)NN * DD;
    float* score2 = R1 + (size_t)2 * NN * DD;
    float* fd2 = R2;                        // agg1 dead after elu->bf16
    float* logits = score1;

    // ---- CSR (shared by both layers) ----
    hipMemsetAsync(deg, 0, (size_t)(NN + 1) * 4, stream);
    count_deg_kernel<<<(EE + 255) / 256, 256, 0, stream>>>(dst, deg, EE);
    exscan_kernel<<<1, 1024, 0, stream>>>(deg, rowptr, NN);
    hipMemcpyAsync(cursor, rowptr, (size_t)(NN + 1) * 4, hipMemcpyDeviceToDevice, stream);
    scatter_edges_kernel<<<(EE + 255) / 256, 256, 0, stream>>>(dst, cursor, eidx, EE);

    // ---- Layer 1 projections (WMMA bf16) ----
    f32_to_bf16_kernel<<<(NN * INF_ + 255) / 256, 256, 0, stream>>>(feat, bfA, (long long)NN * INF_);
    transpose_to_bf16_kernel<<<(INF_ * INF_ + 255) / 256, 256, 0, stream>>>(W1s, w1s_t, INF_, INF_);
    transpose_to_bf16_kernel<<<(INF_ * INF_ + 255) / 256, 256, 0, stream>>>(W1d, w1d_t, INF_, INF_);
    transpose_to_bf16_kernel<<<(INF_ * DD + 255) / 256, 256, 0, stream>>>(W2s, w2s_t, INF_, DD);
    transpose_to_bf16_kernel<<<(INF_ * DD + 255) / 256, 256, 0, stream>>>(W2d, w2d_t, INF_, DD);

    dim3 g1(NN / 16, INF_ / 64);
    wmma_gemm_bf16_kernel<<<g1, 32, 0, stream>>>(bfA, w1s_t, b1s, fs1, INF_, INF_);
    wmma_gemm_bf16_kernel<<<g1, 32, 0, stream>>>(bfA, w1d_t, b1d, fd1, INF_, INF_);

    // ---- Layer 1 edge softmax + aggregate ----
    hipMemsetAsync(msum, 0, (size_t)NN * HH * 4, stream);
    hipMemsetAsync(ssum, 0, (size_t)NN * HH * 4, stream);
    long long EH1 = (long long)EE * HH;
    edge_score_kernel<<<(unsigned)((EH1 + 255) / 256), 256, 0, stream>>>(
        fs1, fd1, src, dst, attn1, score1, msum, EH1, HH, DD);
    edge_exp_kernel<<<(unsigned)((EH1 + 255) / 256), 256, 0, stream>>>(
        score1, dst, msum, ssum, EH1, HH);
    aggregate_kernel<<<NN, HH * DD, 0, stream>>>(score1, ssum, fs1, src, rowptr, eidx, agg1, HH, DD);
    elu_to_bf16_kernel<<<(NN * INF_ + 255) / 256, 256, 0, stream>>>(agg1, bfA, (long long)NN * INF_);

    // ---- Layer 2 projections (WMMA bf16) ----
    dim3 g2(NN / 16, DD / 64);
    wmma_gemm_bf16_kernel<<<g2, 32, 0, stream>>>(bfA, w2s_t, b2s, fs2, INF_, DD);
    wmma_gemm_bf16_kernel<<<g2, 32, 0, stream>>>(bfA, w2d_t, b2d, fd2, INF_, DD);

    // ---- Layer 2 edge softmax + aggregate ----
    hipMemsetAsync(msum, 0, (size_t)NN * 4, stream);
    hipMemsetAsync(ssum, 0, (size_t)NN * 4, stream);
    long long EH2 = (long long)EE;
    edge_score_kernel<<<(unsigned)((EH2 + 255) / 256), 256, 0, stream>>>(
        fs2, fd2, src, dst, attn2, score2, msum, EH2, 1, DD);
    edge_exp_kernel<<<(unsigned)((EH2 + 255) / 256), 256, 0, stream>>>(
        score2, dst, msum, ssum, EH2, 1);
    aggregate_kernel<<<NN, DD, 0, stream>>>(score2, ssum, fs2, src, rowptr, eidx, h2, 1, DD);
    elu_inplace_kernel<<<(NN * DD + 255) / 256, 256, 0, stream>>>(h2, (long long)NN * DD);

    // ---- Set2Set ----
    hipMemsetAsync(q_star, 0, 512, stream);
    hipMemsetAsync(hsbuf, 0, 3 * 64 * 4, stream);
    hipMemsetAsync(csbuf, 0, 3 * 64 * 4, stream);
    for (int it = 0; it < 3; ++it) {
        hipMemsetAsync(rv, 0, 256, stream);
        hipMemsetAsync(gmax, 0, 4, stream);
        hipMemsetAsync(gsum, 0, 4, stream);
        lstm_kernel<<<1, 256, 0, stream>>>(q_star, hsbuf, csbuf,
            lW[0][0], lW[0][1], lW[0][2], lW[0][3],
            lW[1][0], lW[1][1], lW[1][2], lW[1][3],
            lW[2][0], lW[2][1], lW[2][2], lW[2][3], qv);
        s2s_logits_kernel<<<(NN + 255) / 256, 256, 0, stream>>>(h2, qv, logits, gmax, NN);
        s2s_exp_kernel<<<(NN + 255) / 256, 256, 0, stream>>>(logits, gmax, gsum, NN);
        s2s_reduce_kernel<<<64, 256, 0, stream>>>(h2, logits, rv, NN);
        s2s_qstar_kernel<<<1, 128, 0, stream>>>(qv, rv, gsum, q_star);
    }
    copy_out_kernel<<<1, 128, 0, stream>>>(q_star, (float*)d_out, out_size);
}
